// QuantizedLinear_58463094833737
// MI455X (gfx1250) — matrix operand order repaired and checked by reading, not compiled, not run on hardware
//
#include <hip/hip_runtime.h>

// QuantizedLinear: y = x @ ((qw - zp) * scale).T + bias
//  x  [16, 8192] f32, qw [8192, 8192] f32 (integer values 0..255), scale/zp scalar, bias [8192]
// Memory-bound on the 256 MB qw stream (~11.5 us at 23.3 TB/s). Strategy:
//  - qw is exact in bf16 (integers < 256); x split into bf16 hi+lo -> 2x
//    v_wmma_f32_16x16x32_bf16 per K-step gives ~f32 accuracy at HBM-bound cost.
//  - qw loads + y stores are NON-TEMPORAL so the 256 MB one-shot stream does
//    not evict the hot 512 KB x working set from the 192 MB L2 (x is re-read
//    ~512x from L2; losing it would double HBM traffic).
//  - zero_point handled exactly: y = s*(x@qw^T) - s*zp*rowsum(x) + bias.
//  - Block = 8 waves, one 16x16 output tile per block (BATCH=16 == tile M),
//    waves split K, conflict-free LDS reduction of partial f32 C tiles.

#define BATCH 16
#define IN_F  8192
#define OUT_F 8192

typedef __attribute__((ext_vector_type(16))) __bf16 v16bf;
typedef __attribute__((ext_vector_type(8)))  float  v8f;
typedef __attribute__((ext_vector_type(4)))  float  v4f;

// ---------------------------------------------------------------------------
// Kernel 1: S[b] = sum_k x[b,k]  (for the zero-point correction term)
// ---------------------------------------------------------------------------
__global__ __launch_bounds__(256) void qlin_rowsum16(const float* __restrict__ x,
                                                     float* __restrict__ S) {
    __shared__ float red[256];
    const int b = blockIdx.x;
    const float* row = x + (size_t)b * IN_F;
    float s = 0.f;
    for (int k = threadIdx.x; k < IN_F; k += 256) s += row[k];
    red[threadIdx.x] = s;
    __syncthreads();
    #pragma unroll
    for (int off = 128; off > 0; off >>= 1) {
        if (threadIdx.x < off) red[threadIdx.x] += red[threadIdx.x + off];
        __syncthreads();
    }
    if (threadIdx.x == 0) S[b] = red[0];
}

// ---------------------------------------------------------------------------
// Kernel 2: main WMMA GEMM
// ---------------------------------------------------------------------------
__global__ __launch_bounds__(256) void qlin_wmma(const float* __restrict__ x,
                                                 const float* __restrict__ qw,
                                                 const float* __restrict__ scale,
                                                 const float* __restrict__ zp,
                                                 const float* __restrict__ bias,
                                                 const float* __restrict__ S,
                                                 float* __restrict__ y) {
    __shared__ float lds_c[8 * 256];           // 8 waves x 256 f32 partial tiles

    const int t    = threadIdx.x;
    const int lane = t & 31;
    const int wave = t >> 5;
    const int m    = lane & 15;                // A row / B col owned by this lane
    const int sel  = lane >> 4;                // lane half selector
    const int n0   = blockIdx.x * 16;          // output-feature tile base

    const float* xrow = x  + (size_t)m * IN_F;
    const float* qrow = qw + (size_t)(n0 + m) * IN_F;

    v8f c = {0.f, 0.f, 0.f, 0.f, 0.f, 0.f, 0.f, 0.f};

    const int kbeg = wave * (IN_F / 8);        // 1024-wide K slice per wave
    const int kend = kbeg + (IN_F / 8);

    #pragma unroll 2
    for (int k = kbeg; k < kend; k += 32) {
        // A fragment (16x32 bf16): lane holds row m, halves j -> K k+16*sel+j
        // x is hot in L2 (regular temporal loads).
        const float* ap = xrow + k + 16 * sel;
        v4f a0 = *(const v4f*)(ap);
        v4f a1 = *(const v4f*)(ap + 4);
        v4f a2 = *(const v4f*)(ap + 8);
        v4f a3 = *(const v4f*)(ap + 12);
        // B fragment (32x16 bf16): lane holds col n0+m, halves j -> K k+16*sel+j
        // qw is a one-shot 256 MB stream: non-temporal so it can't evict x.
        const float* qp = qrow + k + 16 * sel;
        v4f b0 = __builtin_nontemporal_load((const v4f*)(qp));
        v4f b1 = __builtin_nontemporal_load((const v4f*)(qp + 4));
        v4f b2 = __builtin_nontemporal_load((const v4f*)(qp + 8));
        v4f b3 = __builtin_nontemporal_load((const v4f*)(qp + 12));
        // speculative prefetch 4 K-steps ahead (GL2 fill) on the HBM stream
        __builtin_prefetch(qp + 128, 0, 0);

        float av[16];
        #pragma unroll
        for (int i = 0; i < 4; ++i) {
            av[i]      = a0[i];
            av[4 + i]  = a1[i];
            av[8 + i]  = a2[i];
            av[12 + i] = a3[i];
        }

        v16bf ahi, alo, bv;
        #pragma unroll
        for (int i = 0; i < 4; ++i) {
            bv[i]      = (__bf16)b0[i];        // exact: qw is integer < 256
            bv[4 + i]  = (__bf16)b1[i];
            bv[8 + i]  = (__bf16)b2[i];
            bv[12 + i] = (__bf16)b3[i];
        }
        #pragma unroll
        for (int i = 0; i < 16; ++i) {
            __bf16 h = (__bf16)av[i];          // hi part
            ahi[i] = h;
            alo[i] = (__bf16)(av[i] - (float)h); // residual -> ~f32 accuracy
        }

        c = __builtin_amdgcn_wmma_f32_16x16x32_bf16(false, ahi, false, bv,
                                                    (short)0, c, false, false);
        c = __builtin_amdgcn_wmma_f32_16x16x32_bf16(false, alo, false, bv,
                                                    (short)0, c, false, false);
    }

    // Reduce the 8 per-wave partial C tiles through LDS.
    // Index e = r*32 + lane: stores and loads are bank-conflict-free.
    #pragma unroll
    for (int r = 0; r < 8; ++r) lds_c[wave * 256 + r * 32 + lane] = c[r];
    __syncthreads();

    float acc = 0.f;
    #pragma unroll
    for (int w = 0; w < 8; ++w) acc += lds_c[w * 256 + t];

    // Decode e = t = r*32 + l:  l = t&31, r = t>>5;
    // C/D layout: M = r + 8*(l>=16), N = n0 + (l&15)
    const int l = t & 31;
    const int r = t >> 5;
    const int M = r + 8 * (l >> 4);
    const int N = n0 + (l & 15);

    const float sc = scale[0];
    const float z  = zp[0];
    const float out = sc * acc - sc * z * S[M] + bias[N];
    // y is write-once: non-temporal store, 16 consecutive floats per half-wave.
    __builtin_nontemporal_store(out, y + (size_t)M * OUT_F + N);
}

// ---------------------------------------------------------------------------
// Launch
// ---------------------------------------------------------------------------
extern "C" void kernel_launch(void* const* d_in, const int* in_sizes, int n_in,
                              void* d_out, int out_size, void* d_ws, size_t ws_size,
                              hipStream_t stream) {
    const float* x     = (const float*)d_in[0];
    const float* qw    = (const float*)d_in[1];
    const float* scale = (const float*)d_in[2];
    const float* zp    = (const float*)d_in[3];
    const float* bias  = (const float*)d_in[4];
    float*       y     = (float*)d_out;
    float*       S     = (float*)d_ws;   // 16 floats of scratch

    qlin_rowsum16<<<BATCH, 256, 0, stream>>>(x, S);
    qlin_wmma<<<OUT_F / 16, 256, 0, stream>>>(x, qw, scale, zp, bias, S, y);
}
